// TrSelfAttention_with_ln_res_69295002353947
// MI455X (gfx1250) — matrix operand order repaired and checked by reading, not compiled, not run on hardware
//
#include <hip/hip_runtime.h>

typedef __attribute__((ext_vector_type(16))) _Float16 v16h;
typedef __attribute__((ext_vector_type(8)))  float    v8f;

namespace {
constexpr int BB   = 16;    // batch
constexpr int T    = 36;    // timesteps
constexpr int TP   = 48;    // T padded to 3 WMMA tiles
constexpr int NN   = 207;   // nodes
constexpr int D    = 64;    // model dim
constexpr int H    = 8;     // heads
constexpr int DPH  = 8;     // dim per head
constexpr int C    = 64;    // conv channels
constexpr int KC   = 32;    // conv K padded (24 -> 32)
constexpr int BLK  = 256;   // 8 waves (wave32)
constexpr float SCALE = 0.125f;   // 1/sqrt(64)
constexpr float EPS   = 1e-5f;

// LDS budget (must match carve below)
constexpr int N_FLOATS = TP*D         // Vh (padded rows -> unconditional WMMA stores)
                       + TP*TP        // scores
                       + TP*D         // ctx / outB (padded rows)
                       + 6*64         // biases, gamma, beta
                       + 2*48;        // mu, rstd
constexpr int N_HALVES = 2*(T*D)      // QinH, KinH
                       + TP*D         // VinH (padded)
                       + 2*(D*D)      // WvH, WoH
                       + 2*(C*KC)     // wqp, wkp (zero-padded K)
                       + 2*(TP*KC)    // Aq, Ak im2col (padded)
                       + 2*(TP*C)     // Qh, Kh
                       + TP*D;        // ctxh
constexpr size_t SMEM_BYTES = (size_t)N_FLOATS*4 + (size_t)N_HALVES*2;  // ~98 KB

union HV { _Float16 s[16]; v16h v; };
} // namespace

__global__ __launch_bounds__(BLK)
void fused_tsa_ln_kernel(const float* __restrict__ Xg, const float* __restrict__ Qg,
                         const float* __restrict__ Kg, const float* __restrict__ Vg,
                         const float* __restrict__ Wv, const float* __restrict__ bv,
                         const float* __restrict__ Wo, const float* __restrict__ bo,
                         const float* __restrict__ cqw, const float* __restrict__ cqb,
                         const float* __restrict__ ckw, const float* __restrict__ ckb,
                         const float* __restrict__ gam, const float* __restrict__ bet,
                         float* __restrict__ Og)
{
  extern __shared__ char smem_raw[];
  float* fp = (float*)smem_raw;
  float* Vh  = fp; fp += TP*D;       // rows 36..47 are write-only padding
  float* S   = fp; fp += TP*TP;
  float* ctx = fp; fp += TP*D;       // later reused as outB; rows 36..47 padding
  float* bqs = fp; fp += 64;
  float* bks = fp; fp += 64;
  float* bvs = fp; fp += 64;
  float* bos = fp; fp += 64;
  float* gms = fp; fp += 64;
  float* bts = fp; fp += 64;
  float* muA = fp; fp += 48;
  float* rsA = fp; fp += 48;
  _Float16* hp   = (_Float16*)fp;
  _Float16* QinH = hp; hp += T*D;
  _Float16* KinH = hp; hp += T*D;
  _Float16* VinH = hp; hp += TP*D;
  _Float16* WvH  = hp; hp += D*D;
  _Float16* WoH  = hp; hp += D*D;
  _Float16* wqp  = hp; hp += C*KC;
  _Float16* wkp  = hp; hp += C*KC;
  _Float16* Aq   = hp; hp += TP*KC;
  _Float16* Ak   = hp; hp += TP*KC;
  _Float16* Qh   = hp; hp += TP*C;
  _Float16* Kh   = hp; hp += TP*C;
  _Float16* ctxh = hp; hp += TP*D;

  const int tid  = threadIdx.x;
  const int lane = tid & 31;
  const int wv   = tid >> 5;       // wave id (uniform per wave)
  const int mrow = lane & 15;
  const int hi   = lane >> 4;      // lane half 0/1

  const int bn = blockIdx.x;
  const int bi = bn / NN;
  const int ni = bn - bi * NN;
  const size_t rowbase = ((size_t)bi * T * NN + ni) * D;  // (bi, t=0, ni, 0)
  const size_t tstride = (size_t)NN * D;

  // ---------------- stage inputs & weights ----------------
  for (int idx = tid; idx < T*D; idx += BLK) {
    int t = idx / D, j = idx - t*D;
    size_t g = rowbase + (size_t)t * tstride + j;
    QinH[idx] = (_Float16)Qg[g];
    KinH[idx] = (_Float16)Kg[g];
    VinH[idx] = (_Float16)Vg[g];
  }
  for (int idx = T*D + tid; idx < TP*D; idx += BLK) VinH[idx] = (_Float16)0.f;
  for (int idx = tid; idx < D*D; idx += BLK) {
    WvH[idx] = (_Float16)Wv[idx];
    WoH[idx] = (_Float16)Wo[idx];
  }
  for (int idx = tid; idx < C*KC; idx += BLK) {   // conv weights, K padded 24->32
    int c = idx / KC, k = idx - c*KC;
    wqp[idx] = (k < 24) ? (_Float16)cqw[c*24 + k] : (_Float16)0.f;
    wkp[idx] = (k < 24) ? (_Float16)ckw[c*24 + k] : (_Float16)0.f;
  }
  for (int idx = tid; idx < TP*KC; idx += BLK) {  // zero im2col (pads stay zero)
    Aq[idx] = (_Float16)0.f;
    Ak[idx] = (_Float16)0.f;
  }
  if (tid < 64) {
    bqs[tid] = cqb[tid]; bks[tid] = ckb[tid];
    bvs[tid] = bv[tid];  bos[tid] = bo[tid];
    gms[tid] = gam[tid]; bts[tid] = bet[tid];
  }
  // prefetch residual input X (needed only at the end)
  if (tid < T*4) {
    int t = tid >> 2, ch = tid & 3;
    __builtin_prefetch(&Xg[rowbase + (size_t)t * tstride + ch*16], 0, 0);
  }
  __syncthreads();

  // ---------------- value projection via WMMA: Vh = V @ Wv^T + bv ----------------
  for (int tl = wv; tl < 12; tl += BLK/32) {
    int mt = tl >> 2, nt = tl & 3;
    v8f acc = {0.f,0.f,0.f,0.f,0.f,0.f,0.f,0.f};
    const int arow = (mt*16 + mrow) * D;
    const int brow = (nt*16 + mrow) * D;       // B[K=i][N=j] = Wv[j][i]
    for (int cb = 0; cb < D; cb += 32) {
      HV a, b;
      #pragma unroll
      for (int e = 0; e < 16; ++e) {
        int kA = cb + e + (hi ? 8 : 0) + ((e >= 8) ? 8 : 0);
        a.s[e] = VinH[arow + kA];
        int kB = cb + e + (hi ? 8 : 0) + ((e >= 8) ? 8 : 0);
        b.s[e] = WvH[brow + kB];
      }
      acc = __builtin_amdgcn_wmma_f32_16x16x32_f16(
                false, a.v, false, b.v, (short)0, acc, false, false);
    }
    #pragma unroll
    for (int j = 0; j < 8; ++j) {              // unconditional: Vh has pad rows
      int row = mt*16 + j + hi*8;
      int col = nt*16 + mrow;
      Vh[row*D + col] = acc[j] + bvs[col];
    }
  }
  __syncthreads();

  // causal 16x16 tile list (kt <= qt)
  const int qts[6] = {0,1,1,2,2,2};
  const int kts[6] = {0,0,1,0,1,2};

  // ---------------- per-head attention ----------------
  for (int h = 0; h < H; ++h) {
    // im2col: A[t][k=i*3+tau] = in[t-2+tau][h*8+i]  (pads already zero)
    for (int idx = tid; idx < T*24; idx += BLK) {
      int t = idx / 24, k = idx - t*24;
      int i = k / 3, tau = k - i*3;
      int ts = t - 2 + tau;
      _Float16 zq = (_Float16)0.f, zk = (_Float16)0.f;
      if (ts >= 0) {
        zq = QinH[ts*D + h*DPH + i];
        zk = KinH[ts*D + h*DPH + i];
      }
      Aq[t*KC + k] = zq;
      Ak[t*KC + k] = zk;
    }
    __syncthreads();

    // conv as GEMM via WMMA: Qh = Aq @ wq^T + bq ; Kh = Ak @ wk^T + bk
    // 24 tile tasks (12 per tensor), one K=32 WMMA each, all 8 waves busy
    for (int tl = wv; tl < 24; tl += BLK/32) {
      const int which = (tl >= 12) ? 1 : 0;
      const int tile  = tl - which*12;
      const int mt = tile >> 2, nt = tile & 3;
      const _Float16* Ab = which ? Ak  : Aq;
      const _Float16* Bw = which ? wkp : wqp;
      const float*    bb = which ? bks : bqs;
      _Float16*      Dst = which ? Kh  : Qh;
      v8f acc = {0.f,0.f,0.f,0.f,0.f,0.f,0.f,0.f};
      HV a, b;
      const int arow = (mt*16 + mrow) * KC;
      const int brow = (nt*16 + mrow) * KC;    // B[K=k][N=c] = w[c][k]
      #pragma unroll
      for (int e = 0; e < 16; ++e) {
        int kA = e + (hi ? 8 : 0) + ((e >= 8) ? 8 : 0);
        a.s[e] = Ab[arow + kA];
        int kB = e + (hi ? 8 : 0) + ((e >= 8) ? 8 : 0);
        b.s[e] = Bw[brow + kB];
      }
      acc = __builtin_amdgcn_wmma_f32_16x16x32_f16(
                false, a.v, false, b.v, (short)0, acc, false, false);
      #pragma unroll
      for (int j = 0; j < 8; ++j) {
        int row = mt*16 + j + hi*8;
        int col = nt*16 + mrow;
        Dst[row*C + col] = (_Float16)(acc[j] + bb[col]);
      }
    }
    __syncthreads();

    // scores S = (Qh · Kh^T) * scale via WMMA, one causal tile per wave
    for (int tl = wv; tl < 6; tl += BLK/32) {
      int qt = qts[tl], kt = kts[tl];
      v8f acc = {0.f,0.f,0.f,0.f,0.f,0.f,0.f,0.f};
      const int arow = (qt*16 + mrow) * C;
      const int brow = (kt*16 + mrow) * C;
      for (int cb = 0; cb < C; cb += 32) {
        HV a, b;
        #pragma unroll
        for (int e = 0; e < 16; ++e) {
          int kA = cb + e + (hi ? 8 : 0) + ((e >= 8) ? 8 : 0);
          a.s[e] = Qh[arow + kA];
          int kB = cb + e + (hi ? 8 : 0) + ((e >= 8) ? 8 : 0);
          b.s[e] = Kh[brow + kB];
        }
        acc = __builtin_amdgcn_wmma_f32_16x16x32_f16(
                  false, a.v, false, b.v, (short)0, acc, false, false);
      }
      #pragma unroll
      for (int j = 0; j < 8; ++j) {
        int q = qt*16 + j + hi*8;
        int k = kt*16 + mrow;
        S[q*TP + k] = acc[j] * SCALE;
      }
    }
    __syncthreads();

    // causal softmax, one wave per row with shuffle-butterfly reductions
    for (int q = wv; q < T; q += BLK/32) {
      const int k0 = lane, k1 = lane + 32;
      const bool v0 = (k0 <= q);
      const bool v1 = (k1 <= q);       // q < T <= 36 so k1 < T is implied
      float x0 = v0 ? S[q*TP + k0] : -3.4e38f;
      float x1 = v1 ? S[q*TP + k1] : -3.4e38f;
      float m = fmaxf(x0, x1);
      #pragma unroll
      for (int off = 16; off; off >>= 1) m = fmaxf(m, __shfl_xor(m, off));
      float e0 = v0 ? __expf(x0 - m) : 0.f;
      float e1 = v1 ? __expf(x1 - m) : 0.f;
      float s = e0 + e1;
      #pragma unroll
      for (int off = 16; off; off >>= 1) s += __shfl_xor(s, off);
      float inv = 1.0f / s;
      if (v0) S[q*TP + k0] = e0 * inv;
      if (v1) S[q*TP + k1] = e1 * inv;
    }
    __syncthreads();

    // ctx[t, h*8+e] = sum_k attn[t,k] * Vh[k, h*8+e]  (tiny, f32 VALU)
    for (int idx = tid; idx < T*DPH; idx += BLK) {
      int t = idx / DPH, e = idx - t*DPH;
      float s = 0.f;
      const float* vcol = &Vh[h*DPH + e];
      for (int k = 0; k <= t; ++k) s += S[t*TP + k] * vcol[k*D];
      ctx[t*D + h*DPH + e] = s;
    }
    __syncthreads();
  }

  // ---------------- output projection via WMMA ----------------
  for (int idx = tid; idx < TP*D; idx += BLK) {
    int t = idx / D;
    ctxh[idx] = (t < T) ? (_Float16)ctx[idx] : (_Float16)0.f;
  }
  __syncthreads();

  float* outB = ctx;   // f32 ctx no longer needed; has pad rows 36..47
  for (int tl = wv; tl < 12; tl += BLK/32) {
    int mt = tl >> 2, nt = tl & 3;
    v8f acc = {0.f,0.f,0.f,0.f,0.f,0.f,0.f,0.f};
    const int arow = (mt*16 + mrow) * D;
    const int brow = (nt*16 + mrow) * D;     // B[K=c][N=j] = Wo[j][c]
    for (int cb = 0; cb < D; cb += 32) {
      HV a, b;
      #pragma unroll
      for (int e = 0; e < 16; ++e) {
        int kA = cb + e + (hi ? 8 : 0) + ((e >= 8) ? 8 : 0);
        a.s[e] = ctxh[arow + kA];
        int kB = cb + e + (hi ? 8 : 0) + ((e >= 8) ? 8 : 0);
        b.s[e] = WoH[brow + kB];
      }
      acc = __builtin_amdgcn_wmma_f32_16x16x32_f16(
                false, a.v, false, b.v, (short)0, acc, false, false);
    }
    #pragma unroll
    for (int j = 0; j < 8; ++j) {            // unconditional: outB has pad rows
      int row = mt*16 + j + hi*8;
      outB[row*D + nt*16 + mrow] = acc[j];
    }
  }
  __syncthreads();

  // ---------------- bias + residual + LayerNorm ----------------
  for (int idx = tid; idx < T*D; idx += BLK) {
    int t = idx / D, j = idx - t*D;
    size_t g = rowbase + (size_t)t * tstride + j;
    outB[idx] = outB[idx] + bos[j] + Xg[g];
  }
  __syncthreads();
  for (int q = wv; q < T; q += BLK/32) {     // wave-per-row mean/var
    float x0 = outB[q*D + lane];
    float x1 = outB[q*D + lane + 32];
    float s = x0 + x1, s2 = x0*x0 + x1*x1;
    #pragma unroll
    for (int off = 16; off; off >>= 1) {
      s  += __shfl_xor(s,  off);
      s2 += __shfl_xor(s2, off);
    }
    if (lane == 0) {
      float m   = s * (1.0f / D);
      float var = s2 * (1.0f / D) - m*m;
      muA[q] = m;
      rsA[q] = rsqrtf(var + EPS);
    }
  }
  __syncthreads();
  for (int idx = tid; idx < T*D; idx += BLK) {
    int t = idx / D, j = idx - t*D;
    size_t g = rowbase + (size_t)t * tstride + j;
    Og[g] = (outB[idx] - muA[t]) * rsA[t] * gms[j] + bts[j];
  }
}

extern "C" void kernel_launch(void* const* d_in, const int* in_sizes, int n_in,
                              void* d_out, int out_size, void* d_ws, size_t ws_size,
                              hipStream_t stream) {
  (void)in_sizes; (void)n_in; (void)out_size; (void)d_ws; (void)ws_size;
  const float* X   = (const float*)d_in[0];
  const float* Q   = (const float*)d_in[1];
  const float* K   = (const float*)d_in[2];
  const float* V   = (const float*)d_in[3];
  const float* Wv  = (const float*)d_in[4];
  const float* bv  = (const float*)d_in[5];
  const float* Wo  = (const float*)d_in[6];
  const float* bo  = (const float*)d_in[7];
  const float* cqw = (const float*)d_in[8];
  const float* cqb = (const float*)d_in[9];
  const float* ckw = (const float*)d_in[10];
  const float* ckb = (const float*)d_in[11];
  const float* gam = (const float*)d_in[12];
  const float* bet = (const float*)d_in[13];
  float* out = (float*)d_out;

  dim3 grid(BB * NN);   // 3312 workgroups: one per (batch, node)
  dim3 block(BLK);
  fused_tsa_ln_kernel<<<grid, block, SMEM_BYTES, stream>>>(
      X, Q, K, V, Wv, bv, Wo, bo, cqw, cqb, ckw, ckb, gam, bet, out);
}
